// MultiheadAttention_37580963840511
// MI455X (gfx1250) — matrix operand order repaired and checked by reading, not compiled, not run on hardware
//
#include <hip/hip_runtime.h>
#include <hip/hip_bf16.h>
#include <math.h>
#include <stdint.h>

// ---------------------------------------------------------------------------
// MI455X (gfx1250) multi-head attention, wave32 + V_WMMA_F32_16X16X32_F16.
//
// B=2, S=2048, HID=512, H=8, D=64.  M = B*S = 4096 rows.
// ws layout (bytes):
//   Xb  f16 [4096][512]        @ 0          (4,194,304)
//   Wb  f16 [1536][512]        @ 4,194,304  (1,572,864)
//   Q   f16 [2][8][2048][64]   @ 5,767,168  (4,194,304)   (pre-scaled by 0.125)
//   K   f16 [2][8][2048][64]   @ 9,961,472  (4,194,304)
//   Vt  f16 [2][8][64][2048]   @ 14,155,776 (4,194,304)   (transposed)
// ---------------------------------------------------------------------------

typedef __attribute__((ext_vector_type(16))) _Float16 v16h;
typedef __attribute__((ext_vector_type(8)))  _Float16 v8h;
typedef __attribute__((ext_vector_type(8)))  float    v8f;
typedef __attribute__((ext_vector_type(4)))  int      v4i;

#define WMMA16(a, b, c) \
  __builtin_amdgcn_wmma_f32_16x16x32_f16(false, (a), false, (b), (short)0, (c), false, false)

#if __has_builtin(__builtin_amdgcn_global_load_async_to_lds_b128) && \
    __has_builtin(__builtin_amdgcn_s_wait_asynccnt)
#define HAVE_ASYNC_LDS 1
#else
#define HAVE_ASYNC_LDS 0
#endif

static __device__ __forceinline__ v16h hcat(v8h lo, v8h hi) {
  v16h r;
#pragma unroll
  for (int i = 0; i < 8; ++i) { r[i] = lo[i]; r[i + 8] = hi[i]; }
  return r;
}

// Generic -> hardware address-space v4i pointers via integer round-trip
// (as(3) pointers are 32-bit; low 32 bits of a generic LDS address are the
//  in-aperture offset per the flat-aperture rules).
#define AS1V4(p) ((__attribute__((address_space(1))) v4i*)(uintptr_t)(p))
#define AS3V4(p) ((__attribute__((address_space(3))) v4i*)(uint32_t)(uintptr_t)(p))

// One 16B async (or sync-fallback) global->LDS copy per call.
static __device__ __forceinline__ void stage16(const _Float16* g, _Float16* l) {
#if HAVE_ASYNC_LDS
  __builtin_amdgcn_global_load_async_to_lds_b128(AS1V4(g), AS3V4(l), 0, 0);
#else
  *(v8h*)l = *(const v8h*)g;
#endif
}

// ---------------- Kernel 0: fp32 -> f16 convert for x and W ----------------
__global__ void cvt_f32_to_f16(const float* __restrict__ x,
                               const float* __restrict__ w,
                               _Float16* __restrict__ xb,
                               _Float16* __restrict__ wb) {
  const int nx = 4096 * 512;
  const int nw = 1536 * 512;
  int i = blockIdx.x * blockDim.x + threadIdx.x;
  int stride = gridDim.x * blockDim.x;
  for (int j = i; j < nx; j += stride) xb[j] = (_Float16)x[j];
  for (int j = i; j < nw; j += stride) wb[j] = (_Float16)w[j];
}

// ---------------- Kernel 1: QKV = X @ W^T, scatter into Q/K/Vt -------------
// One wave computes a 16(M) x 64(N) strip, register double-buffered so the
// k-step (i+1) loads are in flight during step i's 4 WMMAs.
__global__ void qkv_gemm(const _Float16* __restrict__ Xb,
                         const _Float16* __restrict__ Wb,
                         _Float16* __restrict__ Qo,
                         _Float16* __restrict__ Ko,
                         _Float16* __restrict__ Vt) {
  const int wave = (blockIdx.x * blockDim.x + threadIdx.x) >> 5;
  const int lane = threadIdx.x & 31;
  const int lrow = lane & 15;
  const bool hi  = lane >= 16;

  const int nStrips = 1536 / 64;              // 24
  const int m0 = (wave / nStrips) * 16;       // row tile (b*s)
  const int n0 = (wave % nStrips) * 64;       // feature strip

  v8f c[4] = {v8f{}, v8f{}, v8f{}, v8f{}};

  const _Float16* xrow = Xb + (size_t)(m0 + lrow) * 512;
  const _Float16* wr[4];
#pragma unroll
  for (int t = 0; t < 4; ++t)
    wr[t] = Wb + (size_t)(n0 + t * 16 + lrow) * 512 + (hi ? 16 : 0);
  const int eo = hi ? 16 : 0;                 // f16-element offset for A halves

  v16h a_cur = hcat(*(const v8h*)(xrow + eo), *(const v8h*)(xrow + eo + 8));
  v16h b_cur[4];
#pragma unroll
  for (int t = 0; t < 4; ++t) b_cur[t] = *(const v16h*)(wr[t]);

#pragma unroll 4
  for (int k0 = 0; k0 < 512; k0 += 32) {
    const int k1 = k0 + 32;
    v16h a_nxt = a_cur;
    v16h b_nxt[4];
#pragma unroll
    for (int t = 0; t < 4; ++t) b_nxt[t] = b_cur[t];
    if (k1 < 512) {
      a_nxt = hcat(*(const v8h*)(xrow + k1 + eo),
                   *(const v8h*)(xrow + k1 + eo + 8));
#pragma unroll
      for (int t = 0; t < 4; ++t) b_nxt[t] = *(const v16h*)(wr[t] + k1);
    }
#pragma unroll
    for (int t = 0; t < 4; ++t) c[t] = WMMA16(a_cur, b_cur[t], c[t]);
    a_cur = a_nxt;
#pragma unroll
    for (int t = 0; t < 4; ++t) b_cur[t] = b_nxt[t];
  }

  // Scatter: feature n -> head hh = n/192, r = n%192; r<64:Q, <128:K, else V^T.
#pragma unroll
  for (int t = 0; t < 4; ++t) {
    const int n  = n0 + t * 16 + lrow;        // C tile: N = lane%16
    const int hh = n / 192;
    const int r  = n % 192;
#pragma unroll
    for (int v = 0; v < 8; ++v) {
      const int srow = m0 + v + (hi ? 8 : 0); // C tile: M = v (+8 upper half)
      const int bb = srow >> 11;
      const int ss = srow & 2047;
      const float val = c[t][v];
      if (r < 64) {
        Qo[(((size_t)(bb * 8 + hh) * 2048) + ss) * 64 + r] = (_Float16)(val * 0.125f);
      } else if (r < 128) {
        Ko[(((size_t)(bb * 8 + hh) * 2048) + ss) * 64 + (r - 64)] = (_Float16)val;
      } else {
        Vt[((size_t)(bb * 8 + hh) * 64 + (r - 128)) * 2048 + ss] = (_Float16)val;
      }
    }
  }
}

// ---------------- Kernel 2: cooperative flash attention --------------------
// One 256-thread block (8 waves) owns one (b,h) and 8 q-tiles.  K (32x64) and
// V^T (64x32) tiles are staged into DOUBLE-BUFFERED LDS with the gfx1250
// async global->LDS engine: stage i+1's loads are issued before computing
// stage i, and `s_wait_asynccnt 2` drains exactly stage i (async loads
// complete in order; the 2 newest per-thread loads are stage i+1's).  This
// overlaps L2->LDS movement with the 8 WMMAs + softmax of each 32-key step,
// and cuts K/V L2 read traffic 8x vs. per-wave loading (~1 GB -> ~128 MB).
__global__ void __launch_bounds__(256)
flash_attn(const _Float16* __restrict__ Q,
           const _Float16* __restrict__ K,
           const _Float16* __restrict__ Vt,
           float* __restrict__ out) {
  __shared__ _Float16 Kl[2][32][72];   // 2 x 4,608 B (row pad 64->72)
  __shared__ _Float16 Vl[2][64][40];   // 2 x 5,120 B (row pad 32->40)

  const int tid    = threadIdx.x;
  const int waveid = tid >> 5;
  const int lane   = tid & 31;
  const int lrow   = lane & 15;
  const bool hi    = lane >= 16;
  const int eo     = hi ? 8 : 0;

  const int bh   = blockIdx.x >> 4;           // 0..15 = b*8+h
  const int qgrp = blockIdx.x & 15;           // group of 8 q-tiles
  const int q0   = (qgrp * 8 + waveid) * 16;  // this wave's query base

  const _Float16* Qh = Q  + (size_t)bh * 2048 * 64;
  const _Float16* Kh = K  + (size_t)bh * 2048 * 64;
  const _Float16* Vh = Vt + (size_t)bh * 64 * 2048;

  // Q^T B-operands, loaded once from global (temperature pre-folded).
  const _Float16* qrow = Qh + (size_t)(q0 + lrow) * 64;
  const v16h qb0 = hcat(*(const v8h*)(qrow + eo), *(const v8h*)(qrow + eo + 16));        // dd 0..31
  const v16h qb1 = hcat(*(const v8h*)(qrow + 32 + eo), *(const v8h*)(qrow + 48 + eo));   // dd 32..63

  // Cooperative staging: 256 threads x (16B K + 16B V) per 32-key step.
  const int krow = tid >> 3, kc8 = (tid & 7) * 8;   // K tile: 32 x 64
  const int vrow = tid >> 2, vc8 = (tid & 3) * 8;   // V tile: 64 x 32

  v8f o[4] = {v8f{}, v8f{}, v8f{}, v8f{}};
  float runM = -__builtin_inff();
  float runL = 0.0f;

  const int NSTEP = 2048 / 32;                // 64

  // Prologue: stage step 0 into buffer 0.
  stage16(Kh + (size_t)krow * 64 + kc8,        &Kl[0][krow][kc8]);
  stage16(Vh + (size_t)vrow * 2048 + vc8,      &Vl[0][vrow][vc8]);

  for (int it = 0; it < NSTEP; ++it) {
    const int cur = it & 1;
    const int k0  = it * 32;

    // Issue next step's async loads into the other buffer, then drain the
    // current step (leave the 2 newest per-thread async loads outstanding).
    if (it + 1 < NSTEP) {
      const int nk0 = k0 + 32, nxt = cur ^ 1;
      stage16(Kh + (size_t)(nk0 + krow) * 64 + kc8,   &Kl[nxt][krow][kc8]);
      stage16(Vh + (size_t)vrow * 2048 + nk0 + vc8,   &Vl[nxt][vrow][vc8]);
#if HAVE_ASYNC_LDS
      __builtin_amdgcn_s_wait_asynccnt(2);
#endif
    } else {
#if HAVE_ASYNC_LDS
      __builtin_amdgcn_s_wait_asynccnt(0);
#endif
    }
    __syncthreads();   // current-step tiles visible to all 8 waves

    // ---- S^T = K_tile x Q^T (two 16-row tiles, Kdim = 64 as 2x32) ----
    const v16h a00 = hcat(*(const v8h*)&Kl[cur][lrow][eo],
                          *(const v8h*)&Kl[cur][lrow][eo + 16]);
    const v16h a01 = hcat(*(const v8h*)&Kl[cur][lrow][32 + eo],
                          *(const v8h*)&Kl[cur][lrow][48 + eo]);
    const v16h a10 = hcat(*(const v8h*)&Kl[cur][lrow + 16][eo],
                          *(const v8h*)&Kl[cur][lrow + 16][eo + 16]);
    const v16h a11 = hcat(*(const v8h*)&Kl[cur][lrow + 16][32 + eo],
                          *(const v8h*)&Kl[cur][lrow + 16][48 + eo]);
    // V B-operands issued early so their DS reads overlap the softmax VALU.
    v16h vb[4];
#pragma unroll
    for (int t = 0; t < 4; ++t) {
      const int vr = t * 16 + lrow;
      vb[t] = hcat(*(const v8h*)&Vl[cur][vr][eo],
                   *(const v8h*)&Vl[cur][vr][eo + 16]);
    }

    v8f s0 = {}; v8f s1 = {};
    s0 = WMMA16(a00, qb0, s0);
    s0 = WMMA16(a01, qb1, s0);
    s1 = WMMA16(a10, qb0, s1);
    s1 = WMMA16(a11, qb1, s1);

    // ---- online softmax over these 32 keys (per-lane + one shfl_xor) ----
    float mloc = s0[0];
#pragma unroll
    for (int v = 0; v < 8; ++v) { mloc = fmaxf(mloc, s0[v]); mloc = fmaxf(mloc, s1[v]); }
    mloc = fmaxf(mloc, __shfl_xor(mloc, 16, 32));
    const float newM  = fmaxf(runM, mloc);
    const float alpha = __expf(runM - newM);

    float p0[8], p1[8];
    float rs = 0.0f;
#pragma unroll
    for (int v = 0; v < 8; ++v) {
      p0[v] = __expf(s0[v] - newM);
      p1[v] = __expf(s1[v] - newM);
      rs += p0[v] + p1[v];
    }
    rs += __shfl_xor(rs, 16, 32);
    runL = runL * alpha + rs;
    runM = newM;

    // P packs directly into the f16 A-operand layout (no lane shuffles).
    v16h pa;
#pragma unroll
    for (int v = 0; v < 8; ++v) { pa[v] = (_Float16)p0[v]; pa[8 + v] = (_Float16)p1[v]; }

    // Per-q rescale broadcast: o-tile VGPR v holds q = v (+8 upper half).
    float sc[8];
#pragma unroll
    for (int v = 0; v < 8; ++v) sc[v] = __shfl(alpha, hi ? (v + 8) : v, 32);

    // ---- O = diag(alpha)*O + P x V ----
#pragma unroll
    for (int t = 0; t < 4; ++t) {
      v8f ot = o[t];
#pragma unroll
      for (int v = 0; v < 8; ++v) ot[v] *= sc[v];
      o[t] = WMMA16(pa, vb[t], ot);
    }

    __syncthreads();   // readers done before this buffer is overwritten (it+2)
  }

  // Finalize: divide by running denominator, write out[b][s][h*64+dcol].
  const float inv = 1.0f / runL;
  float fin[8];
#pragma unroll
  for (int v = 0; v < 8; ++v) fin[v] = __shfl(inv, hi ? (v + 8) : v, 32);

  const int bb = bh >> 3;
  const int hh = bh & 7;
#pragma unroll
  for (int t = 0; t < 4; ++t) {
    const int dcol = t * 16 + lrow;
#pragma unroll
    for (int v = 0; v < 8; ++v) {
      const int s = q0 + v + (hi ? 8 : 0);
      out[((size_t)(bb * 2048 + s)) * 512 + hh * 64 + dcol] = o[t][v] * fin[v];
    }
  }
}

// ---------------------------------------------------------------------------
extern "C" void kernel_launch(void* const* d_in, const int* in_sizes, int n_in,
                              void* d_out, int out_size, void* d_ws, size_t ws_size,
                              hipStream_t stream) {
  const float* x = (const float*)d_in[0];       // [2, 2048, 512] fp32
  const float* w = (const float*)d_in[1];       // [1536, 512]   fp32
  float* out = (float*)d_out;                   // [2, 2048, 512] fp32

  char* ws = (char*)d_ws;
  _Float16* Xb = (_Float16*)(ws);
  _Float16* Wb = (_Float16*)(ws + 4194304);
  _Float16* Qb = (_Float16*)(ws + 5767168);
  _Float16* Kb = (_Float16*)(ws + 9961472);
  _Float16* Vt = (_Float16*)(ws + 14155776);

  cvt_f32_to_f16<<<1024, 256, 0, stream>>>(x, w, Xb, Wb);

  // 4096/16 M-tiles * 1536/64 N-strips = 6144 waves; 8 waves (256 thr) / block.
  qkv_gemm<<<768, 256, 0, stream>>>(Xb, Wb, Qb, Kb, Vt);

  // 16 (b,h) * 16 q-groups = 256 blocks; each block = 8 waves, 8 q-tiles.
  flash_attn<<<256, 256, 0, stream>>>(Qb, Kb, Vt, out);
}
